// LGBlock_33535104647236
// MI455X (gfx1250) — compile-verified, matrix operand order checked by reading, not run on hardware
//
#include <hip/hip_runtime.h>

// ============================================================================
// LGBlock on MI455X (gfx1250, wave32, WMMA, async LDS DMA everywhere).
//
//   0. cvt kernels: cat16 = f16 concat([xq,xs]); Wt = f16 transposed weights
//   1. gemm_wmma<0>: full_qkv16[b,2560,2304] (f16) = cat16 @ qkv_w + b
//                    + transposed V buffer Vt[b,h,64,2560] (f16)
//   2. attn_wmma<false>: cross attn, Q=rows[512,2560), KV=rows[0,2560)
//   3. attn_wmma<true> : self  attn, Q=KV=rows[0,512), scrambled epilogue
//   4. gemm_wmma<1>: out = concat16 @ proj_w + proj_b, split-scatter (f32)
//
// All staged operands are f16 and layout-matched, so every LDS fill in the
// hot loops is a 16B-aligned GLOBAL_LOAD_ASYNC_TO_LDS_B128 (ASYNCcnt DMA,
// no VGPR round-trip), double-buffered: issue chunk i+1, compute chunk i,
// one barrier per iteration. Matrix math: v_wmma_f32_16x16x32_f16.
// Attention is flash-style (online softmax, 32-row KV chunks).
// ============================================================================

typedef __attribute__((ext_vector_type(16))) _Float16 v16h;
typedef __attribute__((ext_vector_type(8)))  float    v8f;
typedef __attribute__((ext_vector_type(4)))  int      int4v;

constexpr int BS   = 2;
constexpr int P    = 512;
constexpr int NQ   = 2048;
constexpr int D    = 768;
constexpr int H    = 12;
constexpr int HD   = 64;
constexpr int ROWS = P + NQ;        // 2560 rows per batch
constexpr int TOTR = BS * ROWS;     // 5120
constexpr int QKVN = 3 * D;         // 2304

#if defined(__has_builtin)
#if __has_builtin(__builtin_amdgcn_global_load_async_to_lds_b128)
#define USE_ASYNC_LDS 1
#endif
#endif
#ifndef USE_ASYNC_LDS
#define USE_ASYNC_LDS 0
#endif

#if USE_ASYNC_LDS
// Signature (from compiler diagnostic): (int4 addrspace(1)*, int4 addrspace(3)*, imm, cpol)
typedef __attribute__((address_space(1))) int4v* gptr128;
typedef __attribute__((address_space(3))) int4v* lptr128;
#endif

// 16-byte global -> LDS copy (per lane). Async DMA when available.
__device__ __forceinline__ void cp_16B(_Float16* l, const _Float16* g) {
#if USE_ASYNC_LDS
  __builtin_amdgcn_global_load_async_to_lds_b128(
      (gptr128)(_Float16*)g, (lptr128)l, 0, 0);
#else
  *(uint4*)l = *(const uint4*)g;
#endif
}
__device__ __forceinline__ void wait_async0() {
#if USE_ASYNC_LDS
  asm volatile("s_wait_asynccnt 0" ::: "memory");
#endif
}

union FragU { v16h v; uint4 q[2]; unsigned u[8]; _Float16 h[16]; };

// ISA 16-bit A-matrix 16x32 K mapping: VGPR v, lane-half -> first K of pair.
__device__ __forceinline__ int kmap(int v, int half) {
  return ((v & 3) << 1) + (half << 3) + ((v >> 2) << 4);
}

// Load a 16x32 f16 fragment from LDS (row-major, `stride` halfs per row).
__device__ __forceinline__ v16h ld_frag(const _Float16* base, int stride) {
  const int L = threadIdx.x & 31;
  const int m = L & 15, half = L >> 4;
  FragU f;
#pragma unroll
  for (int v = 0; v < 8; ++v)
    f.u[v] = *(const unsigned*)(base + m * stride + kmap(v, half));
  return f.v;
}

__device__ __forceinline__ v8f wmma16(v16h a, v16h b, v8f c) {
  return __builtin_amdgcn_wmma_f32_16x16x32_f16(
      false, a, false, b, (short)0, c, false, false);
}

__device__ __forceinline__ float rowmax16(float x) {
#pragma unroll
  for (int off = 1; off < 16; off <<= 1) x = fmaxf(x, __shfl_xor(x, off, 32));
  return x;
}
__device__ __forceinline__ float rowsum16(float x) {
#pragma unroll
  for (int off = 1; off < 16; off <<= 1) x += __shfl_xor(x, off, 32);
  return x;
}

// ============================================================================
// Pre-conversion kernels (one-shot, ~38MB traffic, negligible vs pipeline).
// ============================================================================
// cat16[b,2560,768] = f16(concat([xq, xs], axis=1))
__global__ void __launch_bounds__(256)
cvt_cat_kernel(const float* __restrict__ xq, const float* __restrict__ xs,
               _Float16* __restrict__ dst) {
  int e = blockIdx.x * 256 + threadIdx.x;  // e < TOTR*D, exact grid
  int R = e / D, k = e - R * D;
  int b = R / ROWS, r = R - b * ROWS;
  float v = (r < NQ) ? xq[((size_t)b * NQ + r) * D + k]
                     : xs[((size_t)b * P + (r - NQ)) * D + k];
  dst[e] = (_Float16)v;
}
// Wt[n,768] = f16(W[768,n]^T)  (transpose so B-tiles are DMA-contiguous)
__global__ void __launch_bounds__(256)
cvt_wt_kernel(const float* __restrict__ W, _Float16* __restrict__ Wt, int N) {
  int e = blockIdx.x * 256 + threadIdx.x;  // e < 768*N, exact grid
  int k = e / N, n = e - k * N;
  Wt[(size_t)n * D + k] = (_Float16)W[e];
}

// ============================================================================
// 128x128-tile GEMM: C = A16[M,768] @ Wt[N,768]^T + bias. f16 WMMA, f32 acc.
// Double-buffered async staging; 8 waves, each 32x64 (2x4 WMMA tiles).
// MODE 0: writes f16 full_qkv + transposed per-head Vt.
// MODE 1: writes f32 d_out with the output-split scatter.
// ============================================================================
template <int MODE>
__global__ void __launch_bounds__(256)
gemm_wmma(const _Float16* __restrict__ A16, const _Float16* __restrict__ Wt,
          const float* __restrict__ bias, float* __restrict__ outf,
          _Float16* __restrict__ out16, _Float16* __restrict__ vt, int N) {
  __shared__ _Float16 sA[2][128 * 32];  // [row][k]
  __shared__ _Float16 sB[2][128 * 32];  // [n][k]

  const int tid = threadIdx.x;
  const int L = tid & 31, wave = tid >> 5;
  const int wm = wave & 3, wn = wave >> 2;
  const int rowBase = blockIdx.y * 128;
  const int colBase = blockIdx.x * 128;

  auto stage = [&](int kb, int buf) {
#pragma unroll
    for (int t = 0; t < 2; ++t) {  // 128 rows x 4 segs of 16B each side
      int idx = t * 256 + tid;
      int r = idx >> 2, s8 = (idx & 3) * 8;
      cp_16B(&sA[buf][r * 32 + s8], A16 + (size_t)(rowBase + r) * D + kb + s8);
      cp_16B(&sB[buf][r * 32 + s8], Wt + (size_t)(colBase + r) * D + kb + s8);
    }
  };

  v8f acc[2][4] = {};
  stage(0, 0);

  for (int kb = 0; kb < D; kb += 32) {
    const int cur = (kb >> 5) & 1;
    wait_async0();    // my DMA for `cur` complete
    __syncthreads();  // everyone's DMA landed; previous compute finished
    if (kb + 32 < D) stage(kb + 32, cur ^ 1);  // overlap next DMA with math

    v16h a0 = ld_frag(sA[cur] + (wm * 32 + 0) * 32, 32);
    v16h a1 = ld_frag(sA[cur] + (wm * 32 + 16) * 32, 32);
#pragma unroll
    for (int ni = 0; ni < 4; ++ni) {
      v16h bf = ld_frag(sB[cur] + (wn * 64 + ni * 16) * 32, 32);
      acc[0][ni] = wmma16(a0, bf, acc[0][ni]);
      acc[1][ni] = wmma16(a1, bf, acc[1][ni]);
    }
  }

  const int c = L & 15, half = L >> 4;
#pragma unroll
  for (int mi = 0; mi < 2; ++mi) {
#pragma unroll
    for (int ni = 0; ni < 4; ++ni) {
#pragma unroll
      for (int v = 0; v < 8; ++v) {
        int R = rowBase + wm * 32 + mi * 16 + v + 8 * half;
        int col = colBase + wn * 64 + ni * 16 + c;
        float val = acc[mi][ni][v] + bias[col];
        if (MODE == 0) {
          out16[(size_t)R * N + col] = (_Float16)val;
          if (col >= 2 * D) {  // V section: also write transposed per-head Vt
            int cv = col - 2 * D;
            int hh = cv >> 6, dd = cv & 63;
            int b = R / ROWS, rr = R - b * ROWS;
            vt[((size_t)((b * H + hh) * HD + dd)) * ROWS + rr] = (_Float16)val;
          }
        } else {
          int b = R / ROWS, rr = R - b * ROWS;
          size_t dst = (rr < P)
              ? ((size_t)(b * P + rr)) * D + col
              : (size_t)BS * P * D + ((size_t)(b * NQ + (rr - P))) * D + col;
          outf[dst] = val;
        }
      }
    }
  }
}

// ============================================================================
// Flash attention, double-buffered async KV staging.
// grid=(q_len/128, H, BS), 256 threads = 8 waves; wave owns a 16-row Q tile
// and [16x64] f32 accum (4 WMMA tiles). KV chunks of 32 rows.
// ============================================================================
template <bool SELF>
__global__ void __launch_bounds__(256)
attn_wmma(const _Float16* __restrict__ qkv, const _Float16* __restrict__ vt,
          const unsigned char* __restrict__ mask,
          _Float16* __restrict__ outbuf, int q_start, int kv_len) {
  __shared__ _Float16 sK[2][32 * 64];   // [kv][d] row-major
  __shared__ _Float16 sV[2][64 * 32];   // [d][kv] (Vt is pre-transposed)
  __shared__ _Float16 sP[8 * 16 * 32];  // per-wave probs staging

  const int tid = threadIdx.x, L = tid & 31, wave = tid >> 5;
  const int c = L & 15, half = L >> 4;
  const int b = blockIdx.z, h = blockIdx.y;
  const int qrow = q_start + blockIdx.x * 128 + wave * 16;

  const int kRow = tid >> 3, kSeg = tid & 7;  // K: 32 rows x 8 segs of 16B
  const int vRow = tid >> 2, vSeg = tid & 3;  // V: 64 rows x 4 segs of 16B
  const _Float16* gK0 = qkv + ((size_t)b * ROWS + kRow) * QKVN + D + h * HD + kSeg * 8;
  const _Float16* gV0 = vt + ((size_t)((b * H + h) * HD + vRow)) * ROWS + vSeg * 8;

  auto stage = [&](int kc, int buf) {
    cp_16B(&sK[buf][kRow * 64 + kSeg * 8], gK0 + (size_t)kc * QKVN);
    cp_16B(&sV[buf][vRow * 32 + vSeg * 8], gV0 + kc);
  };

  // Q A-fragments direct from global f16: two aligned 16B groups per lane.
  const _Float16* qp = qkv + ((size_t)b * ROWS + qrow + c) * QKVN + h * HD;
  FragU aq[2];
#pragma unroll
  for (int f = 0; f < 2; ++f) {
    aq[f].q[0] = *(const uint4*)(qp + f * 32 + half * 8);
    aq[f].q[1] = *(const uint4*)(qp + f * 32 + half * 8 + 16);
  }

  v8f o[4] = {};
  float mst[8], lst[8];
#pragma unroll
  for (int v = 0; v < 8; ++v) { mst[v] = -1e30f; lst[v] = 0.0f; }

  _Float16* sPw = sP + wave * 512;
  const float scale = 0.125f;  // 1/sqrt(64)

  stage(0, 0);

  for (int kc = 0; kc < kv_len; kc += 32) {
    const int cur = (kc >> 5) & 1;
    wait_async0();
    __syncthreads();
    if (kc + 32 < kv_len) stage(kc + 32, cur ^ 1);

    // scores[16x32] = Q[16x64] @ K^T : 4 WMMA
    v8f s0 = {}, s1 = {};
#pragma unroll
    for (int ks = 0; ks < 2; ++ks) {
      v16h bk0 = ld_frag(sK[cur] + 0 * 16 * 64 + ks * 32, 64);
      v16h bk1 = ld_frag(sK[cur] + 1 * 16 * 64 + ks * 32, 64);
      s0 = wmma16(aq[ks].v, bk0, s0);
      s1 = wmma16(aq[ks].v, bk1, s1);
    }

    const int kv0 = kc + c, kv1 = kc + 16 + c;
    const bool m0 = (kv0 < P) && mask[b * P + kv0];
    const bool m1 = (kv1 < P) && mask[b * P + kv1];

    // Online softmax (-1e30 sentinel; transient all-masked chunks get wiped
    // by the next rescale corr=exp(-1e30-new_max)=0).
#pragma unroll
    for (int v = 0; v < 8; ++v) {
      float x0 = m0 ? -1e30f : s0[v] * scale;
      float x1 = m1 ? -1e30f : s1[v] * scale;
      float mx = rowmax16(fmaxf(x0, x1));
      float nm = fmaxf(mst[v], mx);
      float corr = __expf(mst[v] - nm);
      float p0 = __expf(x0 - nm);
      float p1 = __expf(x1 - nm);
      lst[v] = lst[v] * corr + rowsum16(p0 + p1);
      mst[v] = nm;
      o[0][v] *= corr; o[1][v] *= corr; o[2][v] *= corr; o[3][v] *= corr;
      sPw[(v + 8 * half) * 32 + c]      = (_Float16)p0;  // C->A layout via LDS
      sPw[(v + 8 * half) * 32 + c + 16] = (_Float16)p1;
    }

    asm volatile("s_wait_dscnt 0" ::: "memory");  // within-wave LDS RAW
    v16h ap = ld_frag(sPw, 32);
#pragma unroll
    for (int j = 0; j < 4; ++j) {  // O[16x64] += P[16x32] @ V[32x64]
      v16h bv = ld_frag(sV[cur] + j * 16 * 32, 32);
      o[j] = wmma16(ap, bv, o[j]);
    }
  }

  // Epilogue: O /= l, write f16 into proj input buffer [b, 2560, 768].
#pragma unroll
  for (int v = 0; v < 8; ++v) {
    float invl = 1.0f / lst[v];
    int qg = qrow + v + 8 * half;
#pragma unroll
    for (int j = 0; j < 4; ++j) {
      float val = o[j][v] * invl;
      int g = h * HD + j * 16 + c;
      size_t dst;
      if (SELF) {
        // quirk: [bs,H,d,P].reshape(bs,P,D): f = (h*64+d)*512 + p
        int f = g * P + qg;
        int rr = f / D, cc = f - rr * D;
        dst = ((size_t)b * ROWS + rr) * D + cc;
      } else {
        dst = ((size_t)b * ROWS + qg) * D + g;  // cross rows are 512..2559
      }
      outbuf[dst] = (_Float16)val;
    }
  }
}

// ============================================================================
extern "C" void kernel_launch(void* const* d_in, const int* in_sizes, int n_in,
                              void* d_out, int out_size, void* d_ws,
                              size_t ws_size, hipStream_t stream) {
  (void)in_sizes; (void)n_in; (void)out_size; (void)ws_size;
  const float* xs     = (const float*)d_in[0];
  const float* xq     = (const float*)d_in[1];
  const unsigned char* xs_mask = (const unsigned char*)d_in[2];  // bool array
  const float* qkv_w  = (const float*)d_in[3];
  const float* qkv_b  = (const float*)d_in[4];
  const float* proj_w = (const float*)d_in[5];
  const float* proj_b = (const float*)d_in[6];
  float* out = (float*)d_out;

  // Workspace layout (f16 halfs): ~52MB total.
  _Float16* cat16     = (_Float16*)d_ws;                    // [5120, 768]
  _Float16* qkvwt16   = cat16 + (size_t)TOTR * D;           // [2304, 768]
  _Float16* projwt16  = qkvwt16 + (size_t)QKVN * D;         // [ 768, 768]
  _Float16* fullqkv16 = projwt16 + (size_t)D * D;           // [5120, 2304]
  _Float16* vtbuf     = fullqkv16 + (size_t)TOTR * QKVN;    // [2,12,64,2560]
  _Float16* concat16  = vtbuf + (size_t)BS * H * HD * ROWS; // [5120, 768]

  // 0) one-shot f16 conversions / transposes
  cvt_cat_kernel<<<(TOTR * D) / 256, 256, 0, stream>>>(xq, xs, cat16);
  cvt_wt_kernel<<<(D * QKVN) / 256, 256, 0, stream>>>(qkv_w, qkvwt16, QKVN);
  cvt_wt_kernel<<<(D * D) / 256, 256, 0, stream>>>(proj_w, projwt16, D);
  // 1) QKV GEMM (f16 out + transposed V): 5120x2304x768
  gemm_wmma<0><<<dim3(QKVN / 128, TOTR / 128), 256, 0, stream>>>(
      cat16, qkvwt16, qkv_b, nullptr, fullqkv16, vtbuf, QKVN);
  // 2) cross attention: Q rows [512,2560), KV rows [0,2560)
  attn_wmma<false><<<dim3(NQ / 128, H, BS), 256, 0, stream>>>(
      fullqkv16, vtbuf, xs_mask, concat16, P, ROWS);
  // 3) self attention: Q=KV rows [0,512), scrambled write
  attn_wmma<true><<<dim3(P / 128, H, BS), 256, 0, stream>>>(
      fullqkv16, vtbuf, xs_mask, concat16, 0, P);
  // 4) proj GEMM + output split: 5120x768x768
  gemm_wmma<1><<<dim3(D / 128, TOTR / 128), 256, 0, stream>>>(
      concat16, projwt16, proj_b, out, nullptr, nullptr, D);
}
